// Hash_Mapping_24867860644184
// MI455X (gfx1250) — matrix operand order repaired and checked by reading, not compiled, not run on hardware
//
#include <hip/hip_runtime.h>
#include <hip/hip_bf16.h>

typedef __attribute__((ext_vector_type(16))) _Float16 v16h;
typedef __attribute__((ext_vector_type(8)))  _Float16 v8h;
typedef __attribute__((ext_vector_type(8)))  float    v8f;

#define TBL_SIZE (1u << 19)
#define TBL_MASK (TBL_SIZE - 1u)

__constant__ float c_res[16] = {
    16.f, 24.f, 36.f, 54.f, 81.f, 121.f, 182.f, 273.f,
    410.f, 615.f, 922.f, 1383.f, 2075.f, 3113.f, 4670.f, 7006.f};

union ABFrag { v16h v; v8h h[2]; };

// ---------------------------------------------------------------------------
// Prep: transpose + f32->f16 convert W1 [128,256] -> W1t [256,128],
//                                  W2 [256,512] -> W2t [512,256]
// ---------------------------------------------------------------------------
__global__ __launch_bounds__(256) void prep_w_kernel(
    const float* __restrict__ W1, const float* __restrict__ W2,
    _Float16* __restrict__ W1t, _Float16* __restrict__ W2t) {
  int i = blockIdx.x * blockDim.x + threadIdx.x;
  if (i < 128 * 256) {
    int k = i >> 8, n = i & 255;
    W1t[n * 128 + k] = (_Float16)W1[i];
  }
  int j = i - 128 * 256;
  if (j >= 0 && j < 256 * 512) {
    int k = j >> 9, n = j & 511;
    W2t[n * 256 + k] = (_Float16)W2[j];
  }
}

// ---------------------------------------------------------------------------
// Hash encoding: one thread per (sample b, group g, level l).
// Writes feature pair (2 x f16) packed into one u32 -> fully coalesced.
// ---------------------------------------------------------------------------
__global__ __launch_bounds__(256) void hash_enc_kernel(
    const float* __restrict__ z, const float* __restrict__ tables,
    unsigned* __restrict__ Xpacked) {
  int tid = blockIdx.x * blockDim.x + threadIdx.x;
  int b = tid >> 6;
  int g = (tid >> 4) & 3;
  int l = tid & 15;

  float res = c_res[l];
  float fr[4];
  unsigned fl[4];
#pragma unroll
  for (int d = 0; d < 4; ++d) {
    float x = z[b * 16 + g * 4 + d];
    float s = 1.0f / (1.0f + __expf(-x));      // sigmoid
    float sc = s * (res - 1.0f);
    float f = floorf(sc);
    fr[d] = sc - f;
    fl[d] = (unsigned)(int)f;
  }

  const float2* tp = (const float2*)tables + (size_t)(g * 16 + l) * TBL_SIZE;
  float f0 = 0.0f, f1 = 0.0f;
#pragma unroll
  for (int c = 0; c < 16; ++c) {
    unsigned b0 = c & 1, b1c = (c >> 1) & 1, b2c = (c >> 2) & 1, b3 = (c >> 3) & 1;
    unsigned h = (fl[0] + b0) * 1u
               ^ (fl[1] + b1c) * 2654435761u
               ^ (fl[2] + b2c) * 805459861u
               ^ (fl[3] + b3) * 3674653429u;
    h &= TBL_MASK;
    float w = (b0  ? fr[0] : 1.0f - fr[0])
            * (b1c ? fr[1] : 1.0f - fr[1])
            * (b2c ? fr[2] : 1.0f - fr[2])
            * (b3  ? fr[3] : 1.0f - fr[3]);
    float2 v = tp[h];
    f0 += w * v.x;
    f1 += w * v.y;
  }

  union { _Float16 h[2]; unsigned u; } pk;
  pk.h[0] = (_Float16)f0;
  pk.h[1] = (_Float16)f1;
  Xpacked[tid] = pk.u;   // column = g*32 + l*2 {+0,+1} of X[b,128]
}

// ---------------------------------------------------------------------------
// Fused MLP: out = leaky_relu(X@W1+b1) @ W2 + b2 via v_wmma_f32_16x16x32_f16.
// Block = 8 waves / 64 rows. B-stationary register blocking: each wave holds
// 4 M-tile accumulators and reuses each global B fragment 4x.
// ---------------------------------------------------------------------------
#define BM 64
#define XS_STRIDE 136   // 128 + 8 halves pad -> breaks LDS bank conflicts
#define HS_STRIDE 264   // 256 + 8 halves pad

__global__ __launch_bounds__(256) void mlp_kernel(
    const _Float16* __restrict__ Xf16,
    const _Float16* __restrict__ W1t, const float* __restrict__ b1,
    const _Float16* __restrict__ W2t, const float* __restrict__ b2,
    float* __restrict__ out) {
  __shared__ _Float16 Xs[BM * XS_STRIDE];
  __shared__ _Float16 Hs[BM * HS_STRIDE];

  int tid = threadIdx.x;
  int mbase = blockIdx.x * BM;

  // Stage X tile (64 x 128 f16) into padded LDS, 16B vectors.
  {
    const uint4* src = (const uint4*)(Xf16 + (size_t)mbase * 128);
    for (int i = tid; i < BM * 16; i += 256) {
      int row = i >> 4, seg = i & 15;
      *(uint4*)&Xs[row * XS_STRIDE + seg * 8] = src[(size_t)row * 16 + seg];
    }
  }
  __syncthreads();

  int wave = tid >> 5, lane = tid & 31;
  int lhi = lane >> 4, llo = lane & 15;

  // ---- GEMM1: [64,128] x [128,256], leaky-relu -> Hs ----
  for (int nt = wave; nt < 16; nt += 8) {
    int col = nt * 16 + llo;            // N index (lane)
    v8f acc[4] = {};
#pragma unroll
    for (int kt = 0; kt < 4; ++kt) {
      // B 32x16: lane=column, contiguous K (transposed weights); loaded once,
      // reused for all 4 M-tiles.
      ABFrag bm;
      const v8h* bp = (const v8h*)(W1t + (size_t)col * 128 + kt * 32 + lhi * 16);
      bm.h[0] = bp[0];
      bm.h[1] = bp[1];
#pragma unroll
      for (int mt = 0; mt < 4; ++mt) {
        // A 16x32 as loaded here: lane<16 K {0..15}, lane>=16 {16..31} (matches B)
        ABFrag a;
        int row = mt * 16 + llo;
        a.h[0] = *(const v8h*)&Xs[row * XS_STRIDE + kt * 32 + lhi * 16];
        a.h[1] = *(const v8h*)&Xs[row * XS_STRIDE + kt * 32 + lhi * 16 + 8];
        acc[mt] = __builtin_amdgcn_wmma_f32_16x16x32_f16(
            false, a.v, false, bm.v, (short)0, acc[mt], false, false);
      }
    }
    float bias = b1[col];
#pragma unroll
    for (int mt = 0; mt < 4; ++mt) {
#pragma unroll
      for (int r = 0; r < 8; ++r) {
        float x = acc[mt][r] + bias;
        x = x > 0.0f ? x : 0.01f * x;                  // leaky_relu
        Hs[(mt * 16 + r + lhi * 8) * HS_STRIDE + col] = (_Float16)x;
      }
    }
  }
  __syncthreads();

  // ---- GEMM2: [64,256] x [256,512] + b2 -> out ----
  for (int nt = wave; nt < 32; nt += 8) {
    int col = nt * 16 + llo;
    v8f acc[4] = {};
#pragma unroll
    for (int kt = 0; kt < 8; ++kt) {
      ABFrag bm;
      const v8h* bp = (const v8h*)(W2t + (size_t)col * 256 + kt * 32 + lhi * 16);
      bm.h[0] = bp[0];
      bm.h[1] = bp[1];
#pragma unroll
      for (int mt = 0; mt < 4; ++mt) {
        ABFrag a;
        int row = mt * 16 + llo;
        a.h[0] = *(const v8h*)&Hs[row * HS_STRIDE + kt * 32 + lhi * 16];
        a.h[1] = *(const v8h*)&Hs[row * HS_STRIDE + kt * 32 + lhi * 16 + 8];
        acc[mt] = __builtin_amdgcn_wmma_f32_16x16x32_f16(
            false, a.v, false, bm.v, (short)0, acc[mt], false, false);
      }
    }
    float bias = b2[col];
#pragma unroll
    for (int mt = 0; mt < 4; ++mt) {
#pragma unroll
      for (int r = 0; r < 8; ++r) {
        // f32 C/D layout: VGPR r -> row r (lanes 0-15) / row r+8 (lanes 16-31)
        out[(size_t)(mbase + mt * 16 + r + lhi * 8) * 512 + col] = acc[mt][r] + bias;
      }
    }
  }
}

// ---------------------------------------------------------------------------
extern "C" void kernel_launch(void* const* d_in, const int* in_sizes, int n_in,
                              void* d_out, int out_size, void* d_ws, size_t ws_size,
                              hipStream_t stream) {
  const float* z      = (const float*)d_in[0];
  const float* tables = (const float*)d_in[1];
  const float* W1     = (const float*)d_in[2];
  const float* b1     = (const float*)d_in[3];
  const float* W2     = (const float*)d_in[4];
  const float* b2     = (const float*)d_in[5];
  float* out = (float*)d_out;

  const int B = in_sizes[0] / 16;   // 131072

  // Workspace layout: X f16 [B,128] | W1t f16 [256,128] | W2t f16 [512,256]
  _Float16* Xf16 = (_Float16*)d_ws;
  _Float16* W1t  = (_Float16*)((char*)d_ws + (size_t)B * 128 * sizeof(_Float16));
  _Float16* W2t  = W1t + 256 * 128;

  prep_w_kernel<<<(128 * 256 + 256 * 512 + 255) / 256, 256, 0, stream>>>(
      W1, W2, W1t, W2t);

  hash_enc_kernel<<<(B * 64) / 256, 256, 0, stream>>>(
      z, tables, (unsigned*)Xf16);

  mlp_kernel<<<B / BM, 256, 0, stream>>>(Xf16, W1t, b1, W2t, b2, out);
}
